// WindowTransformerBlock_75350906241266
// MI455X (gfx1250) — matrix operand order repaired and checked by reading, not compiled, not run on hardware
//
#include <hip/hip_runtime.h>
#include <hip/hip_bf16.h>

typedef __attribute__((ext_vector_type(16))) _Float16 v16h;
typedef __attribute__((ext_vector_type(8)))  _Float16 v8h;
typedef __attribute__((ext_vector_type(8)))  float    v8f;

#define BATCH 8
#define DIM   128
#define HSZ   192
#define WSZ   192
#define HEADS 4
#define WS    8
#define SHIFT 4
#define NTOK  64
#define HD    32
#define NWX   24
#define NWIN  576
#define QKVW_ELEMS (3*DIM*DIM)   // 49152
#define PROJW_ELEMS (DIM*DIM)    // 16384

// LDS layout (bytes)
#define OFF_SX   0        // 64x128 f16  (input window; later O)
#define OFF_SQ   16384    // 64x128 f16  (q, pre-scaled)
#define OFF_SK   32768    // 64x128 f16
#define OFF_SVT  49152    // 128x64 f16  (v transposed)
#define OFF_SS   65536    // 4x64x64 f32 (S; P f16 in place; later 64x128 f32 out-stage)
#define OFF_RPB  131072   // 900 f32
#define OFF_QB   134672   // 384 f32
#define OFF_PB   136208   // 128 f32
#define OFF_REG  136720   // 64 i32
#define LDS_BYTES 137216

// ---- WMMA fragment loaders (v_wmma_f32_16x16x32_f16) --------------------
// A operand: lane (hl,r) holds row r, K = hl*16 .. hl*16+15 contiguous
__device__ __forceinline__ v16h frag_a(const _Float16* base, int ld, int r, int hl) {
    return *(const v16h*)(base + r * ld + hl * 16);
}
// B operand: lane (hl,r) holds column r (== row r of row-major B^T source),
// K = hl*16 .. hl*16+15 contiguous -> one 32B load
__device__ __forceinline__ v16h frag_b(const _Float16* base, int ld, int r, int hl) {
    return *(const v16h*)(base + r * ld + hl * 16);
}

__device__ __forceinline__ int region_id(int hh, int ww) {
    int rh = (hh < HSZ - WS) ? 0 : ((hh < HSZ - SHIFT) ? 1 : 2);
    int rw = (ww < WSZ - WS) ? 0 : ((ww < WSZ - SHIFT) ? 1 : 2);
    return rh * 3 + rw;
}

// ---- weight pre-conversion f32 -> f16 into workspace --------------------
__global__ void prep_weights(const float* __restrict__ qkv_w,
                             const float* __restrict__ proj_w,
                             _Float16* __restrict__ wsh) {
    int e = blockIdx.x * 256 + threadIdx.x;
    if (e < QKVW_ELEMS)                     wsh[e] = (_Float16)qkv_w[e];
    else if (e < QKVW_ELEMS + PROJW_ELEMS)  wsh[e] = (_Float16)proj_w[e - QKVW_ELEMS];
}

// ---- fused window-attention kernel: one workgroup per window ------------
__global__ __launch_bounds__(256)
void swin_window_kernel(const float* __restrict__ x,
                        const float* __restrict__ qkv_b,
                        const float* __restrict__ proj_b,
                        const float* __restrict__ rpb,
                        const _Float16* __restrict__ wqkv,   // (384,128) f16
                        const _Float16* __restrict__ wproj,  // (128,128) f16
                        float* __restrict__ out) {
    extern __shared__ __attribute__((aligned(128))) char smem[];
    _Float16* sx   = (_Float16*)(smem + OFF_SX);
    _Float16* sq   = (_Float16*)(smem + OFF_SQ);
    _Float16* sk   = (_Float16*)(smem + OFF_SK);
    _Float16* svT  = (_Float16*)(smem + OFF_SVT);
    float*    sS   = (float*)(smem + OFF_SS);
    float*    sRPB = (float*)(smem + OFF_RPB);
    float*    sQB  = (float*)(smem + OFF_QB);
    float*    sPB  = (float*)(smem + OFF_PB);
    int*      sReg = (int*)(smem + OFF_REG);

    const int tid  = threadIdx.x;
    const int wv   = tid >> 5;
    const int lane = tid & 31;
    const int hl   = lane >> 4;
    const int r16  = lane & 15;

    const int win  = blockIdx.x;
    const int b    = win / NWIN;
    const int wrem = win % NWIN;
    const int wy   = wrem / NWX;
    const int wx   = wrem % NWX;

    // ---- phase 0: gather shifted window + preload tables ----------------
    {
        const float* xb = x + (size_t)b * DIM * HSZ * WSZ;
#pragma unroll 4
        for (int i = 0; i < 32; ++i) {
            int e = i * 256 + tid;
            int c = e >> 6;
            int t = e & 63;
            int hh = (wy * WS + (t >> 3) + SHIFT) % HSZ;
            int ww = (wx * WS + (t & 7) + SHIFT) % WSZ;
            sx[t * DIM + c] = (_Float16)xb[(size_t)c * (HSZ * WSZ) + hh * WSZ + ww];
        }
        for (int e = tid; e < 900; e += 256) sRPB[e] = rpb[e];
        for (int e = tid; e < 384; e += 256) sQB[e] = qkv_b[e];
        if (tid < 128) sPB[tid] = proj_b[tid];
        if (tid < 64)  sReg[tid] = region_id(wy * WS + (tid >> 3), wx * WS + (tid & 7));
    }
    __syncthreads();

    const float scale = 0.17677669529663687f;   // HD^-0.5

    // ---- phase 1: QKV^T tiles: D[m=outcol][n=token] ---------------------
    // Q: outcols 0..127, pre-scaled, packed b128 stores
#pragma unroll
    for (int tt = 0; tt < 4; ++tt) {
        int T = wv + tt * 8;               // 0..31
        int mt = T >> 2, nt = T & 3;
        v8f acc = {};
#pragma unroll
        for (int kb = 0; kb < 4; ++kb) {
            v16h a  = frag_a(wqkv + mt * 16 * DIM + kb * 32, DIM, r16, hl);
            v16h bf = frag_b(sx + kb * 32, DIM, nt * 16 + r16, hl);
            acc = __builtin_amdgcn_wmma_f32_16x16x32_f16(false, a, false, bf,
                                                         (short)0, acc, false, false);
        }
        int tok = nt * 16 + r16;
        v8h pk;
#pragma unroll
        for (int i = 0; i < 8; ++i) {
            int col = mt * 16 + hl * 8 + i;
            pk[i] = (_Float16)((acc[i] + sQB[col]) * scale);
        }
        *(v8h*)(sq + tok * DIM + mt * 16 + hl * 8) = pk;
    }
    // K: outcols 128..255
#pragma unroll
    for (int tt = 0; tt < 4; ++tt) {
        int T = wv + tt * 8;
        int mt = T >> 2, nt = T & 3;
        v8f acc = {};
#pragma unroll
        for (int kb = 0; kb < 4; ++kb) {
            v16h a  = frag_a(wqkv + (128 + mt * 16) * DIM + kb * 32, DIM, r16, hl);
            v16h bf = frag_b(sx + kb * 32, DIM, nt * 16 + r16, hl);
            acc = __builtin_amdgcn_wmma_f32_16x16x32_f16(false, a, false, bf,
                                                         (short)0, acc, false, false);
        }
        int tok = nt * 16 + r16;
        v8h pk;
#pragma unroll
        for (int i = 0; i < 8; ++i) {
            int col = mt * 16 + hl * 8 + i;
            pk[i] = (_Float16)(acc[i] + sQB[128 + col]);
        }
        *(v8h*)(sk + tok * DIM + mt * 16 + hl * 8) = pk;
    }
    // V: outcols 256..383, stored transposed svT[d][t]
#pragma unroll
    for (int tt = 0; tt < 4; ++tt) {
        int T = wv + tt * 8;
        int mt = T >> 2, nt = T & 3;
        v8f acc = {};
#pragma unroll
        for (int kb = 0; kb < 4; ++kb) {
            v16h a  = frag_a(wqkv + (256 + mt * 16) * DIM + kb * 32, DIM, r16, hl);
            v16h bf = frag_b(sx + kb * 32, DIM, nt * 16 + r16, hl);
            acc = __builtin_amdgcn_wmma_f32_16x16x32_f16(false, a, false, bf,
                                                         (short)0, acc, false, false);
        }
        int tok = nt * 16 + r16;
#pragma unroll
        for (int i = 0; i < 8; ++i) {
            int d = mt * 16 + hl * 8 + i;
            svT[d * NTOK + tok] = (_Float16)(acc[i] + sQB[256 + d]);
        }
    }
    __syncthreads();

    // ---- phase 2: S^T tiles: D[m=key j][n=query q] + bias + mask --------
#pragma unroll
    for (int tt = 0; tt < 8; ++tt) {
        int T = wv + tt * 8;               // 0..63
        int head = T >> 4;
        int mt = (T >> 2) & 3;             // key tile
        int nt = T & 3;                    // query tile
        v8f acc = {};
        v16h a  = frag_a(sk + mt * 16 * DIM + head * HD, DIM, r16, hl);
        v16h bf = frag_b(sq + head * HD, DIM, nt * 16 + r16, hl);
        acc = __builtin_amdgcn_wmma_f32_16x16x32_f16(false, a, false, bf,
                                                     (short)0, acc, false, false);
        int qq = nt * 16 + r16;            // query token
        int qy = qq >> 3, qx = qq & 7;
        int regq = sReg[qq];
        v8f vals;
#pragma unroll
        for (int i = 0; i < 8; ++i) {
            int j  = mt * 16 + hl * 8 + i; // key token
            int jy = j >> 3, jx = j & 7;
            int ridx = (qy - jy + WS - 1) * (2 * WS - 1) + (qx - jx + WS - 1);
            float mval = (regq != sReg[j]) ? -100.0f : 0.0f;
            vals[i] = acc[i] + sRPB[ridx * HEADS + head] + mval;
        }
        *(v8f*)(sS + head * NTOK * NTOK + qq * NTOK + mt * 16 + hl * 8) = vals;
    }
    __syncthreads();

    // ---- phase 3: softmax rows; P f16 written in place over f32 row -----
    {
        float* rowp = sS + tid * NTOK;     // tid == head*64 + query
        _Float16* prow = (_Float16*)rowp;  // row stride = 128 halves
        float mx = -3.0e38f;
#pragma unroll 8
        for (int j = 0; j < NTOK; ++j) mx = fmaxf(mx, rowp[j]);
        float sum = 0.0f;
        for (int j = 0; j < NTOK; ++j) {   // half j overlaps float j/2 <= j: safe
            float e = __expf(rowp[j] - mx);
            sum += e;
            prow[j] = (_Float16)e;
        }
        float inv = 1.0f / sum;
#pragma unroll 8
        for (int j = 0; j < NTOK; ++j) prow[j] = (_Float16)((float)prow[j] * inv);
    }
    __syncthreads();

    // ---- phase 4: O^T tiles: D[m=dim][n=token] -> sx (packed) -----------
    const _Float16* Ph = (const _Float16*)sS;  // [head*64+q][128] halves, 64 valid
#pragma unroll
    for (int tt = 0; tt < 4; ++tt) {
        int T = wv + tt * 8;               // 0..31
        int head = T >> 3;
        int mt = (T >> 2) & 1;             // dim tile within head
        int nt = T & 3;                    // token tile
        v8f acc = {};
#pragma unroll
        for (int kb = 0; kb < 2; ++kb) {
            v16h a  = frag_a(svT + (head * HD + mt * 16) * NTOK + kb * 32, NTOK, r16, hl);
            v16h bf = frag_b(Ph + head * NTOK * 128 + kb * 32, 128, nt * 16 + r16, hl);
            acc = __builtin_amdgcn_wmma_f32_16x16x32_f16(false, a, false, bf,
                                                         (short)0, acc, false, false);
        }
        int tok = nt * 16 + r16;
        v8h pk;
#pragma unroll
        for (int i = 0; i < 8; ++i) pk[i] = (_Float16)acc[i];
        *(v8h*)(sx + tok * DIM + head * HD + mt * 16 + hl * 8) = pk;
    }
    __syncthreads();

    // ---- phase 5: proj^T tiles -> f32 stage in LDS (packed) -------------
    float* sF = sS;                         // 64x128 f32 out-stage (32KB)
#pragma unroll
    for (int tt = 0; tt < 4; ++tt) {
        int T = wv + tt * 8;               // 0..31
        int mt = T >> 2, nt = T & 3;
        v8f acc = {};
#pragma unroll
        for (int kb = 0; kb < 4; ++kb) {
            v16h a  = frag_a(wproj + mt * 16 * DIM + kb * 32, DIM, r16, hl);
            v16h bf = frag_b(sx + kb * 32, DIM, nt * 16 + r16, hl);
            acc = __builtin_amdgcn_wmma_f32_16x16x32_f16(false, a, false, bf,
                                                         (short)0, acc, false, false);
        }
        int tok = nt * 16 + r16;
        v8f vals;
#pragma unroll
        for (int i = 0; i < 8; ++i) vals[i] = acc[i] + sPB[mt * 16 + hl * 8 + i];
        *(v8f*)(sF + tok * DIM + mt * 16 + hl * 8) = vals;
    }
    __syncthreads();

    // ---- phase 6: coalesced writeout with window-reverse + un-shift -----
    {
        float* outb = out + (size_t)b * DIM * HSZ * WSZ;
#pragma unroll 4
        for (int i = 0; i < 32; ++i) {
            int e = i * 256 + tid;
            int c = e >> 6;
            int t = e & 63;
            int hh = (wy * WS + (t >> 3) + SHIFT) % HSZ;
            int ww = (wx * WS + (t & 7) + SHIFT) % WSZ;
            outb[(size_t)c * (HSZ * WSZ) + hh * WSZ + ww] = sF[t * DIM + c];
        }
    }
}

extern "C" void kernel_launch(void* const* d_in, const int* in_sizes, int n_in,
                              void* d_out, int out_size, void* d_ws, size_t ws_size,
                              hipStream_t stream) {
    const float* x      = (const float*)d_in[0];
    const float* qkv_w  = (const float*)d_in[1];
    const float* qkv_b  = (const float*)d_in[2];
    const float* proj_w = (const float*)d_in[3];
    const float* proj_b = (const float*)d_in[4];
    const float* rpb    = (const float*)d_in[5];
    float* out          = (float*)d_out;
    _Float16* wsh       = (_Float16*)d_ws;

    prep_weights<<<(QKVW_ELEMS + PROJW_ELEMS + 255) / 256, 256, 0, stream>>>(qkv_w, proj_w, wsh);

    swin_window_kernel<<<BATCH * NWIN, 256, LDS_BYTES, stream>>>(
        x, qkv_b, proj_b, rpb, wsh, wsh + QKVW_ELEMS, out);
}
